// LinearAttention_37512244363786
// MI455X (gfx1250) — matrix operand order repaired and checked by reading, not compiled, not run on hardware
//
#include <hip/hip_runtime.h>
#include <math.h>

typedef __bf16 bf16;
typedef __attribute__((ext_vector_type(16))) __bf16 v16bf;
typedef __attribute__((ext_vector_type(8)))  float  v8f;
typedef __attribute__((ext_vector_type(4)))  unsigned int u32x4;
typedef __attribute__((ext_vector_type(8)))  unsigned int u32x8;

#define B_     16
#define C_     256
#define N_     4096
#define O3_    1536
#define H_     8
#define D_     64
#define HID_   512
#define NKPAD_ 4128   /* 4096 main + 4 mem + 28 zero pad : multiple of 32 */

// ---------- scalar helpers ----------
__device__ __forceinline__ bf16 to_bf16(float f) {
  unsigned u = __builtin_bit_cast(unsigned, f);
  unsigned r = u + 0x7FFFu + ((u >> 16) & 1u);   // round-to-nearest-even
  unsigned short h = (unsigned short)(r >> 16);
  return __builtin_bit_cast(bf16, h);
}

__device__ __forceinline__ v8f v8f_zero() {
  v8f z;
#pragma unroll
  for (int e = 0; e < 8; ++e) z[e] = 0.f;
  return z;
}

// ---------- TDM: 2D tile (bf16) global -> LDS, descriptor per ISA 08 §8.3/8.4 ----------
// tile_x elements contiguous (X), tile_y lines (Y), line stride = stride_elems.
__device__ __forceinline__ void tdm_load_2d_bf16(unsigned lds_off, const void* gaddr,
                                                 unsigned tile_x, unsigned tile_y,
                                                 unsigned long long stride_elems) {
  unsigned long long ga = (unsigned long long)(uintptr_t)gaddr;
  const unsigned tdim = 1u << 20;   // tiles always in-bounds; never trigger OOB fill
  u32x4 g0;
  g0[0] = 1u;                                    // count=1, user descriptor
  g0[1] = lds_off;                               // lds_addr (bytes)
  g0[2] = (unsigned)ga;                          // global_addr[31:0]
  g0[3] = ((unsigned)(ga >> 32) & 0x01FFFFFFu)   // global_addr[56:32]
          | (2u << 30);                          // type=2 ("image")
  u32x8 g1;
  g1[0] = (1u << 16);                            // workgroup_mask=0, data_size=1 (2 bytes)
  g1[1] = (tdim & 0xFFFFu) << 16;                // abar_addr=0 | tensor_dim0[15:0]
  g1[2] = (tdim >> 16) | ((tdim & 0xFFFFu) << 16);        // tensor_dim0[31:16] | tensor_dim1[15:0]
  g1[3] = (tdim >> 16) | ((tile_x & 0xFFFFu) << 16);      // tensor_dim1[31:16] | tile_dim0
  g1[4] = tile_y & 0xFFFFu;                      // tile_dim1 | tile_dim2=0
  g1[5] = (unsigned)stride_elems;                // tensor_dim0_stride[31:0]
  g1[6] = (unsigned)(stride_elems >> 32) & 0xFFFFu;  // d0_stride[47:32] | d1_stride[15:0]=0
  g1[7] = 0u;
  asm volatile("tensor_load_to_lds %0, %1" :: "s"(g0), "s"(g1) : "memory");
}

// ---------- WMMA fragment helpers (ISA 7.12.2 layouts, wave32) ----------
// A: 16x32 bf16, source row-major [M][K] with leading stride lda.
__device__ __forceinline__ v16bf load_frag_a(const bf16* __restrict__ A, int lda) {
  int lane = threadIdx.x & 31;
  int m  = lane & 15;
  int kb = (lane >> 4) << 3;          // 0 or 8
  const bf16* row = A + (size_t)m * lda;
  v16bf f;
#pragma unroll
  for (int i = 0; i < 8; ++i) {
    f[i]     = row[kb + i];
    f[8 + i] = row[kb + 16 + i];
  }
  return f;
}

// B: 32x16 bf16 from N-major storage Bt[n][k] (stride ldb along n).
__device__ __forceinline__ v16bf load_frag_b_nk(const bf16* __restrict__ Bt, int ldb) {
  int lane = threadIdx.x & 31;
  int n  = lane & 15;
  int kb = (lane >> 4) << 3;          // 0 or 8
  const bf16* row = Bt + (size_t)n * ldb + kb;
  v16bf f;
#pragma unroll
  for (int i = 0; i < 8; ++i) { f[i] = row[i]; f[8 + i] = row[16 + i]; }
  return f;
}

__device__ __forceinline__ v8f wmma_bf16(v16bf a, v16bf b, v8f c) {
  return __builtin_amdgcn_wmma_f32_16x16x32_bf16(false, a, false, b, (short)0, c, false, false);
}

// C/D 16x16 f32: VGPR r -> M = r + 8*(lane>=16), N = lane%16
__device__ __forceinline__ void store_tile_f32(float* __restrict__ Dp, int ldd, v8f c) {
  int lane = threadIdx.x & 31;
  int n  = lane & 15;
  int mb = (lane >> 4) << 3;
#pragma unroll
  for (int r = 0; r < 8; ++r) Dp[(size_t)(mb + r) * ldd + n] = c[r];
}

// ---------- K0: f32 -> bf16 weight conversion ----------
__global__ void cvt_f32_bf16(const float* __restrict__ src, bf16* __restrict__ dst, int n) {
  int i = blockIdx.x * 256 + threadIdx.x;
  if (i < n) dst[i] = to_bf16(src[i]);
}

// ---------- K1: rmsnorm over C per position, cast bf16 ----------
__global__ void rmsnorm_in_kernel(const float* __restrict__ x, const float* __restrict__ g,
                                  bf16* __restrict__ xn) {
  int b = blockIdx.y;
  int p = blockIdx.x * 256 + threadIdx.x;
  size_t base = (size_t)b * C_ * N_;
  float ss = 0.f;
  for (int c = 0; c < C_; ++c) {
    float v = x[base + (size_t)c * N_ + p];
    ss += v * v;
  }
  float s = sqrtf((float)C_) / fmaxf(sqrtf(ss), 1e-12f);
  for (int c = 0; c < C_; ++c) {
    float v = x[base + (size_t)c * N_ + p] * s * g[c];
    xn[base + (size_t)c * N_ + p] = to_bf16(v);
  }
}

// ---------- K2: qkv = w_qkv(1536x256) * xn(256x4096) per batch ----------
// A tile (weights) arrives via TDM (double-buffered, TENSORcnt); B tile is a
// cooperative transposed copy (needed for the B-fragment N-major layout).
__global__ void qkv_gemm(const bf16* __restrict__ Wq, const bf16* __restrict__ Xn,
                         float* __restrict__ Qkv) {
  __shared__ bf16 As[2][128][32];   // [buf][m][k], TDM destination
  __shared__ bf16 Bt[128][32];      // [n][k]
  int b  = blockIdx.z;
  int tM = blockIdx.y * 128;
  int tN = blockIdx.x * 128;
  const bf16* xb = Xn + (size_t)b * C_ * N_;
  float*      ob = Qkv + (size_t)b * O3_ * N_;
  int tid  = threadIdx.x;
  int wave = tid >> 5;
  int wm = wave >> 2, wn = wave & 3;
  bool issuer = (wave == 0);        // wave-uniform branch; TDM issues once per wave
  v8f acc[4][2];
#pragma unroll
  for (int i = 0; i < 4; ++i)
#pragma unroll
    for (int j = 0; j < 2; ++j) acc[i][j] = v8f_zero();

  const bf16* wrow = Wq + (size_t)tM * C_;
  if (issuer)
    tdm_load_2d_bf16((unsigned)(uintptr_t)&As[0][0][0], wrow, 32, 128, (unsigned long long)C_);

  for (int k0 = 0; k0 < C_; k0 += 32) {
    int cur = (k0 >> 5) & 1;
    bool more = (k0 + 32) < C_;
    if (issuer && more)
      tdm_load_2d_bf16((unsigned)(uintptr_t)&As[cur ^ 1][0][0], wrow + (k0 + 32),
                       32, 128, (unsigned long long)C_);
    // cooperative transposed B tile
    {
      int n = tid & 127, kb = (tid >> 7) * 16;
#pragma unroll
      for (int i = 0; i < 16; ++i)
        Bt[n][kb + i] = xb[(size_t)(k0 + kb + i) * N_ + tN + n];
    }
    if (issuer) {
      if (more) __builtin_amdgcn_s_wait_tensorcnt(1);   // current tile landed
      else      __builtin_amdgcn_s_wait_tensorcnt(0);
    }
    __syncthreads();
    v16bf af[4], bfv[2];
#pragma unroll
    for (int mi = 0; mi < 4; ++mi) af[mi]  = load_frag_a(&As[cur][wm * 64 + mi * 16][0], 32);
#pragma unroll
    for (int ni = 0; ni < 2; ++ni) bfv[ni] = load_frag_b_nk(&Bt[wn * 32 + ni * 16][0], 32);
#pragma unroll
    for (int mi = 0; mi < 4; ++mi)
#pragma unroll
      for (int ni = 0; ni < 2; ++ni)
        acc[mi][ni] = wmma_bf16(af[mi], bfv[ni], acc[mi][ni]);
    __syncthreads();
  }
#pragma unroll
  for (int mi = 0; mi < 4; ++mi)
#pragma unroll
    for (int ni = 0; ni < 2; ++ni)
      store_tile_f32(&ob[(size_t)(tM + wm * 64 + mi * 16) * N_ + tN + wn * 32 + ni * 16],
                     N_, acc[mi][ni]);
}

// ---------- K3: q softmax over head_dim, * scale, -> bf16 ----------
__global__ void q_softmax(const float* __restrict__ Qkv, bf16* __restrict__ Qb) {
  int b = blockIdx.z, h = blockIdx.y;
  int n = blockIdx.x * 256 + threadIdx.x;
  const float* src = Qkv + ((size_t)b * O3_ + h * D_) * N_ + n;
  float m = -INFINITY;
  for (int d = 0; d < D_; ++d) m = fmaxf(m, src[(size_t)d * N_]);
  float s = 0.f;
  for (int d = 0; d < D_; ++d) s += __expf(src[(size_t)d * N_] - m);
  float inv = 0.125f / s;  // HEAD_DIM^-0.5 folded in
  bf16* dst = Qb + ((size_t)b * HID_ + h * D_) * N_ + n;
  for (int d = 0; d < D_; ++d)
    dst[(size_t)d * N_] = to_bf16(__expf(src[(size_t)d * N_] - m) * inv);
}

// ---------- K4: k softmax over n (4096 + 4 mem), write padded bf16 row ----------
__global__ void k_softmax(const float* __restrict__ Qkv, const float* __restrict__ MemKV,
                          bf16* __restrict__ Kc) {
  __shared__ float red[256];
  int b = blockIdx.y, o2 = blockIdx.x;
  int h = o2 >> 6, d = o2 & 63;
  const float* row = Qkv + ((size_t)b * O3_ + HID_ + o2) * N_;
  float memv[4];
#pragma unroll
  for (int j = 0; j < 4; ++j) memv[j] = MemKV[(h * D_ + d) * 4 + j];  // mem_k (s=0)
  int tid = threadIdx.x;
  float lm = -INFINITY;
  for (int n = tid; n < N_; n += 256) lm = fmaxf(lm, row[n]);
  if (tid < 4) lm = fmaxf(lm, memv[tid]);
  red[tid] = lm; __syncthreads();
  for (int s = 128; s > 0; s >>= 1) {
    if (tid < s) red[tid] = fmaxf(red[tid], red[tid + s]);
    __syncthreads();
  }
  float rmax = red[0]; __syncthreads();
  float ls = 0.f;
  for (int n = tid; n < N_; n += 256) ls += __expf(row[n] - rmax);
  if (tid < 4) ls += __expf(memv[tid] - rmax);
  red[tid] = ls; __syncthreads();
  for (int s = 128; s > 0; s >>= 1) {
    if (tid < s) red[tid] += red[tid + s];
    __syncthreads();
  }
  float inv = 1.f / red[0];
  bf16* dst = Kc + ((size_t)b * HID_ + o2) * NKPAD_;
  for (int n = tid; n < N_; n += 256) dst[n] = to_bf16(__expf(row[n] - rmax) * inv);
  if (tid < 32) {
    float v = (tid < 4) ? __expf(memv[tid] - rmax) * inv : 0.f;
    dst[N_ + tid] = to_bf16(v);
  }
}

// ---------- K5: v concat + mem_v broadcast + zero pad, bf16 ----------
__global__ void v_concat(const float* __restrict__ Qkv, const float* __restrict__ MemKV,
                         bf16* __restrict__ Vc) {
  size_t i = (size_t)blockIdx.x * 256 + threadIdx.x;
  size_t total = (size_t)B_ * HID_ * NKPAD_;
  if (i >= total) return;
  int n = (int)(i % NKPAD_);
  size_t bo = i / NKPAD_;
  int o2 = (int)(bo % HID_);
  int b  = (int)(bo / HID_);
  float v;
  if (n < N_) {
    v = Qkv[((size_t)b * O3_ + 2 * HID_ + o2) * N_ + n];
  } else if (n < N_ + 4) {
    int h = o2 >> 6, d = o2 & 63;
    v = MemKV[2048 + (h * D_ + d) * 4 + (n - N_)];  // mem_v (s=1)
  } else {
    v = 0.f;
  }
  Vc[i] = to_bf16(v);
}

// ---------- K6: context[d][e] = sum_n k[d,n] v[e,n]; store transposed bf16 ----------
__global__ void context_gemm(const bf16* __restrict__ Kc, const bf16* __restrict__ Vc,
                             bf16* __restrict__ Ct) {
  int b = blockIdx.y, h = blockIdx.x;
  const bf16* kb = Kc + ((size_t)b * HID_ + h * D_) * NKPAD_;
  const bf16* vb = Vc + ((size_t)b * HID_ + h * D_) * NKPAD_;
  int wave = threadIdx.x >> 5;
  int mi = wave & 3;          // d tile
  int nj = (wave >> 2) * 2;   // e tiles nj, nj+1
  v8f acc[2];
  acc[0] = v8f_zero(); acc[1] = v8f_zero();
  for (int kk = 0; kk < NKPAD_; kk += 32) {
    v16bf a = load_frag_a(kb + (size_t)(mi * 16) * NKPAD_ + kk, NKPAD_);
#pragma unroll
    for (int j = 0; j < 2; ++j) {
      v16bf bb = load_frag_b_nk(vb + (size_t)((nj + j) * 16) * NKPAD_ + kk, NKPAD_);
      acc[j] = wmma_bf16(a, bb, acc[j]);
    }
  }
  bf16* ct = Ct + (size_t)(b * H_ + h) * D_ * D_;   // [e][d]
  int lane = threadIdx.x & 31;
  int nn = lane & 15, mb = (lane >> 4) << 3;
#pragma unroll
  for (int j = 0; j < 2; ++j)
#pragma unroll
    for (int r = 0; r < 8; ++r)
      ct[((nj + j) * 16 + nn) * D_ + mi * 16 + mb + r] = to_bf16(acc[j][r]);
}

// ---------- K7: out[e][n] = sum_d ct[e][d] q[d][n] -> bf16 ----------
__global__ void attn_out_gemm(const bf16* __restrict__ Ct, const bf16* __restrict__ Qb,
                              bf16* __restrict__ Ab) {
  __shared__ bf16 qT[128][64];   // [n][d]
  int b = blockIdx.z, h = blockIdx.y;
  int tN = blockIdx.x * 128;
  const bf16* qsrc = Qb + ((size_t)b * HID_ + h * D_) * N_;
  const bf16* ct   = Ct + (size_t)(b * H_ + h) * D_ * D_;
  int tid = threadIdx.x;
#pragma unroll
  for (int i = 0; i < 32; ++i) {
    int idx = tid + i * 256;
    int d = idx >> 7, n = idx & 127;
    qT[n][d] = qsrc[(size_t)d * N_ + tN + n];
  }
  __syncthreads();
  int wave = tid >> 5;
  int mi = wave & 3;          // e tile
  int nb = (wave >> 2) * 4;   // n tiles nb..nb+3
  v8f acc[4];
#pragma unroll
  for (int j = 0; j < 4; ++j) acc[j] = v8f_zero();
  for (int kk = 0; kk < D_; kk += 32) {
    v16bf a = load_frag_a(ct + (size_t)(mi * 16) * D_ + kk, D_);
#pragma unroll
    for (int j = 0; j < 4; ++j) {
      v16bf bb = load_frag_b_nk(&qT[(nb + j) * 16][kk], 64);
      acc[j] = wmma_bf16(a, bb, acc[j]);
    }
  }
  bf16* dst = Ab + ((size_t)b * HID_ + h * D_) * N_ + tN;
  int lane = tid & 31, nn = lane & 15, mb = (lane >> 4) << 3;
#pragma unroll
  for (int j = 0; j < 4; ++j)
#pragma unroll
    for (int r = 0; r < 8; ++r)
      dst[(size_t)(mi * 16 + mb + r) * N_ + (nb + j) * 16 + nn] = to_bf16(acc[j][r]);
}

// ---------- K8: out = w_out*attn + b_out, then rmsnorm over C ----------
__global__ void out_proj_rmsnorm(const bf16* __restrict__ Wo, const bf16* __restrict__ Ab,
                                 const float* __restrict__ Bias, const float* __restrict__ Gout,
                                 float* __restrict__ Out) {
  __shared__ bf16 bT[64][128];   // [n][c-chunk]
  __shared__ float ssq[64];
  int b  = blockIdx.y;
  int tN = blockIdx.x * 64;
  int tid = threadIdx.x;
  if (tid < 64) ssq[tid] = 0.f;
  int wave = tid >> 5, lane = tid & 31;
  int nt = wave & 3;                // n tile (16 cols)
  int mg = (wave >> 2) * 128;       // m base (0 or 128)
  v8f acc[8];
#pragma unroll
  for (int i = 0; i < 8; ++i) acc[i] = v8f_zero();
  const bf16* asrc = Ab + (size_t)b * HID_ * N_ + tN;

  for (int c0 = 0; c0 < HID_; c0 += 128) {
    __syncthreads();
#pragma unroll
    for (int i = 0; i < 32; ++i) {
      int idx = tid + i * 256;
      int cc = idx >> 6, n = idx & 63;
      bT[n][cc] = asrc[(size_t)(c0 + cc) * N_ + n];
    }
    __syncthreads();
    for (int kk = 0; kk < 128; kk += 32) {
      v16bf bb = load_frag_b_nk(&bT[nt * 16][kk], 128);
#pragma unroll
      for (int i = 0; i < 8; ++i) {
        v16bf a = load_frag_a(Wo + (size_t)(mg + i * 16) * HID_ + c0 + kk, HID_);
        acc[i] = wmma_bf16(a, bb, acc[i]);
      }
    }
  }
  // epilogue: + bias, column sum-of-squares via LDS atomics, rmsnorm, store
  int nn = lane & 15, mb = (lane >> 4) << 3;
  int col = nt * 16 + nn;
  float partial = 0.f;
#pragma unroll
  for (int i = 0; i < 8; ++i)
#pragma unroll
    for (int r = 0; r < 8; ++r) {
      int m = mg + i * 16 + mb + r;
      float v = acc[i][r] + Bias[m];
      acc[i][r] = v;
      partial += v * v;
    }
  atomicAdd(&ssq[col], partial);
  __syncthreads();
  float scale = sqrtf((float)C_) / fmaxf(sqrtf(ssq[col]), 1e-12f);
  float* outp = Out + (size_t)b * C_ * N_ + tN;
#pragma unroll
  for (int i = 0; i < 8; ++i)
#pragma unroll
    for (int r = 0; r < 8; ++r) {
      int m = mg + i * 16 + mb + r;
      outp[(size_t)m * N_ + col] = acc[i][r] * scale * Gout[m];
    }
}

// ---------- launch ----------
extern "C" void kernel_launch(void* const* d_in, const int* in_sizes, int n_in,
                              void* d_out, int out_size, void* d_ws, size_t ws_size,
                              hipStream_t stream) {
  (void)in_sizes; (void)n_in; (void)out_size; (void)ws_size;
  const float* x      = (const float*)d_in[0];
  const float* g_in   = (const float*)d_in[1];
  const float* w_qkv  = (const float*)d_in[2];
  const float* mem_kv = (const float*)d_in[3];
  const float* w_out  = (const float*)d_in[4];
  const float* b_out  = (const float*)d_in[5];
  const float* g_out  = (const float*)d_in[6];
  float* out = (float*)d_out;

  char* ws = (char*)d_ws;
  size_t off = 0;
  auto alloc = [&](size_t bytes) -> void* {
    void* p = ws + off;
    off += (bytes + 255) & ~(size_t)255;
    return p;
  };
  bf16*  wq_bf  = (bf16*)alloc((size_t)O3_ * C_ * 2);
  bf16*  wo_bf  = (bf16*)alloc((size_t)C_ * HID_ * 2);
  bf16*  xn     = (bf16*)alloc((size_t)B_ * C_ * N_ * 2);
  float* qkv    = (float*)alloc((size_t)B_ * O3_ * N_ * 4);
  bf16*  q_bf   = (bf16*)alloc((size_t)B_ * HID_ * N_ * 2);
  bf16*  k_cat  = (bf16*)alloc((size_t)B_ * HID_ * NKPAD_ * 2);
  bf16*  v_cat  = (bf16*)alloc((size_t)B_ * HID_ * NKPAD_ * 2);
  bf16*  ct     = (bf16*)alloc((size_t)B_ * H_ * D_ * D_ * 2);
  bf16*  a_bf   = (bf16*)alloc((size_t)B_ * HID_ * N_ * 2);

  cvt_f32_bf16<<<(O3_ * C_ + 255) / 256, 256, 0, stream>>>(w_qkv, wq_bf, O3_ * C_);
  cvt_f32_bf16<<<(C_ * HID_ + 255) / 256, 256, 0, stream>>>(w_out, wo_bf, C_ * HID_);

  rmsnorm_in_kernel<<<dim3(N_ / 256, B_), 256, 0, stream>>>(x, g_in, xn);

  qkv_gemm<<<dim3(N_ / 128, O3_ / 128, B_), 256, 0, stream>>>(wq_bf, xn, qkv);

  q_softmax<<<dim3(N_ / 256, H_, B_), 256, 0, stream>>>(qkv, q_bf);
  k_softmax<<<dim3(HID_, B_), 256, 0, stream>>>(qkv, mem_kv, k_cat);
  {
    size_t total = (size_t)B_ * HID_ * NKPAD_;
    v_concat<<<(unsigned)((total + 255) / 256), 256, 0, stream>>>(qkv, mem_kv, v_cat);
  }

  context_gemm<<<dim3(H_, B_), 256, 0, stream>>>(k_cat, v_cat, ct);

  attn_out_gemm<<<dim3(N_ / 128, H_, B_), 256, 0, stream>>>(ct, q_bf, a_bf);

  out_proj_rmsnorm<<<dim3(N_ / 64, B_), 256, 0, stream>>>(wo_bf, a_bf, b_out, g_out, out);
}
